// FlowNet_base_46531675685389
// MI455X (gfx1250) — compile-verified, matrix operand order checked by reading, not run on hardware
//
#include <hip/hip_runtime.h>
#include <hip/hip_bf16.h>
#include <stdint.h>

typedef __attribute__((ext_vector_type(16))) _Float16 v16h;
typedef __attribute__((ext_vector_type(8)))  _Float16 v8h;
typedef __attribute__((ext_vector_type(8)))  float    v8f;

#define HH 7
#define WW 34
#define PIX 238            // 7*34 real pixels
#define SW  36             // padded spatial width  (halo col each side)
#define SLOTS 360          // 10 padded rows * 36   (covers taps of dead lanes too)
#define CIN 512
#define CH 529             // live concat channels
#define CHP 544            // padded channel stride (mult of 32; CHP-inOff == icPad per layer)
#define NSTRIDE ((size_t)SLOTS * CHP)   // NHWC per-batch stride (halves)
#define KFC 125902         // 529*238 (exact fc1 K)
#define KFCP 125920        // padded NCHW batch stride (mult of 32)
#define NB 128             // batch

__device__ __forceinline__ float leakyf(float v) { return v >= 0.f ? v : 0.1f * v; }

// CDNA5 async data-mover: memory -> LDS, 16B per lane, tracked by ASYNCcnt.
__device__ __forceinline__ void async_copy_b128(uint32_t lds_off, const void* gaddr) {
  asm volatile("global_load_async_to_lds_b128 %0, %1, off"
               :: "v"(lds_off), "v"(gaddr) : "memory");
}
__device__ __forceinline__ void wait_asynccnt0() {
  asm volatile("s_wait_asynccnt 0" ::: "memory");
}

__device__ __forceinline__ v16h frag16(v8h lo, v8h hi) {
  return __builtin_shufflevector(lo, hi, 0,1,2,3,4,5,6,7,8,9,10,11,12,13,14,15);
}

// Zero the whole NHWC buffer (halo slots + pad channels stay 0 afterwards).
__global__ void zeroN_kernel(_Float16* __restrict__ xN) {
  const size_t i = (size_t)blockIdx.x * blockDim.x + threadIdx.x;
  const size_t total16 = (size_t)NB * NSTRIDE / 8;   // uint4 = 8 halves
  if (i < total16) ((uint4*)xN)[i] = make_uint4(0, 0, 0, 0);
}

// Weight repack: OIHW f32 -> [kk][oc][icPad] f16, zero-padded to icPad.
__global__ void pack_w_kernel(const float* __restrict__ w, _Float16* __restrict__ dst,
                              int oc, int ic, int icPad) {
  const int idx = blockIdx.x * blockDim.x + threadIdx.x;
  const int total = 9 * oc * icPad;
  if (idx >= total) return;
  const int kk  = idx / (oc * icPad);
  const int r   = idx - kk * oc * icPad;
  const int o   = r / icPad;
  const int icx = r - o * icPad;
  _Float16 v = (_Float16)0.f;
  if (icx < ic) v = (_Float16)w[((size_t)o * ic + icx) * 9 + kk];
  dst[idx] = v;
}

// -------------------------------------------------------------------------
// corr81 + leaky -> channels [448..529) of both layouts. left/right (63 MB
// each) stay resident in the 192 MB L2, so the 81x reuse is L2 traffic.
// -------------------------------------------------------------------------
__global__ void corr81_kernel(const float* __restrict__ left,
                              const float* __restrict__ right,
                              _Float16* __restrict__ xN,   // NHWC padded slots
                              _Float16* __restrict__ xC) { // NCHW, stride KFCP
  const int b = blockIdx.x / 81;
  const int d = blockIdx.x % 81;
  const int dy = d / 9 - 4;
  const int dx = d % 9 - 4;
  const int p = threadIdx.x;
  if (p >= PIX) return;
  const int y = p / WW, xw = p - y * WW;
  const int y2 = y + dy, x2 = xw + dx;
  const bool inb = (y2 >= 0) & (y2 < HH) & (x2 >= 0) & (x2 < WW);
  const float* L = left  + (size_t)b * CIN * PIX + p;
  const float* R = right + (size_t)b * CIN * PIX + (inb ? (y2 * WW + x2) : 0);
  float acc = 0.f;
  if (inb) {
    #pragma unroll 8
    for (int c = 0; c < CIN; ++c)
      acc += L[c * PIX] * R[c * PIX];
  }
  const _Float16 v = (_Float16)leakyf(acc * (1.f / (float)CIN));
  const int slot = (y + 1) * SW + (xw + 1);
  xN[(size_t)b * NSTRIDE + (size_t)slot * CHP + (448 + d)] = v;
  xC[(size_t)b * KFCP + (size_t)(448 + d) * PIX + p] = v;
}

// Zero the NCHW per-batch tail pad [125902..125920).
__global__ void padC_kernel(_Float16* __restrict__ xC) {
  const int idx = blockIdx.x * blockDim.x + threadIdx.x;
  if (idx >= NB * (KFCP - KFC)) return;
  const int b = idx / (KFCP - KFC);
  const int t = idx - b * (KFCP - KFC);
  xC[(size_t)b * KFCP + KFC + t] = (_Float16)0.f;
}

// -------------------------------------------------------------------------
// Implicit-GEMM 3x3 conv on WMMA. Halo-padded NHWC => unconditional B loads
// (2x global_load_b128 per WMMA, zero taps come from memory). Weight tile
// [16 x icPad] per (ky,kx) double-buffered in LDS via the async data mover:
// issue copies for step kk+1, compute step kk, s_wait_asynccnt + barrier.
// -------------------------------------------------------------------------
__global__ void conv3_wmma_kernel(_Float16* __restrict__ xN,
                                  _Float16* __restrict__ xC,
                                  const _Float16* __restrict__ wp, // [kk][oc][icPad]
                                  const float* __restrict__ bias,
                                  int oc, int inOff, int icPad, int outOff) {
  __shared__ __align__(16) _Float16 smA[2][16 * 512];

  const int octiles = oc >> 4;
  int rem = blockIdx.x;
  const int b = rem / (octiles * 2); rem -= b * octiles * 2;
  const int ocT   = rem >> 1;
  const int halfB = rem & 1;

  const int lane = threadIdx.x & 31;
  const int l15  = lane & 15;
  const int hi   = lane >> 4;
  const int pT   = halfB * 8 + (threadIdx.x >> 5);   // pixel tile 0..15 (15 = idle)
  const bool act = (pT < 15);
  const int ocBase = ocT * 16;
  const int p  = pT * 16 + l15;                      // <= 239, taps stay in-bounds
  const int py = p / WW, px = p - py * WW;

  const _Float16* xin = xN + (size_t)b * NSTRIDE;
  const uint32_t smB[2] = { (uint32_t)(uintptr_t)(&smA[0][0]),
                            (uint32_t)(uintptr_t)(&smA[1][0]) };
  const uint32_t tileBytes = (uint32_t)(16 * icPad * 2);

  // prologue: stage kk=0 weight slice
  {
    const char* src = (const char*)(wp + (size_t)ocBase * icPad);
    for (uint32_t ofs = threadIdx.x * 16u; ofs < tileBytes; ofs += 256u * 16u)
      async_copy_b128(smB[0] + ofs, src + ofs);
    wait_asynccnt0();
    __syncthreads();
  }

  v8f acc = {};
  for (int kk = 0; kk < 9; ++kk) {
    const int cur = kk & 1;
    if (kk < 8) {  // overlap: issue next slice while computing this one
      const char* src = (const char*)(wp + ((size_t)(kk + 1) * oc + ocBase) * icPad);
      for (uint32_t ofs = threadIdx.x * 16u; ofs < tileBytes; ofs += 256u * 16u)
        async_copy_b128(smB[cur ^ 1] + ofs, src + ofs);
    }
    if (act) {
      const int ky = kk / 3, kx = kk - ky * 3;
      // halo-padded slot: always valid, halo reads return stored zeros
      const _Float16* bp = xin + (size_t)((py + ky) * SW + (px + kx)) * CHP + inOff;
      const _Float16* ap = &smA[cur][0] + l15 * icPad + hi * 8;
      for (int k0 = 0; k0 < icPad; k0 += 32) {
        const v8h alo = *(const v8h*)(ap + k0);
        const v8h ahi = *(const v8h*)(ap + k0 + 16);
        const v8h blo = *(const v8h*)(bp + k0 + hi * 8);
        const v8h bhi = *(const v8h*)(bp + k0 + 16 + hi * 8);
        acc = __builtin_amdgcn_wmma_f32_16x16x32_f16(
            false, frag16(alo, ahi), false, frag16(blo, bhi),
            (short)0, acc, false, false);
      }
    }
    wait_asynccnt0();
    __syncthreads();
  }

  if (act && p < PIX) {
    #pragma unroll
    for (int j = 0; j < 8; ++j) {
      const int o = ocBase + j + (hi ? 8 : 0);
      const _Float16 v = (_Float16)leakyf(acc[j] + bias[o]);
      xN[(size_t)b * NSTRIDE + (size_t)((py + 1) * SW + (px + 1)) * CHP + (outOff + o)] = v;
      xC[(size_t)b * KFCP + (size_t)(outOff + o) * PIX + p] = v;
    }
  }
}

// -------------------------------------------------------------------------
// WMMA FC: out[m,n] = act(A[m,:K].W[n,:K] + bias[n]). A f16 (row stride lda,
// 32-aligned), W f32 (row stride ldw), vector main loop + one guarded tail.
// -------------------------------------------------------------------------
__global__ void fc_wmma_kernel(const _Float16* __restrict__ A, int lda,
                               const float* __restrict__ W, int ldw,
                               const float* __restrict__ bias,
                               _Float16* __restrict__ outH,
                               int M, int N, int K, int doLeaky) {
  const int lane = threadIdx.x & 31;
  const int l15  = lane & 15;
  const int hi   = lane >> 4;
  const int ntiles = N >> 4;
  const int wid = blockIdx.x * 8 + (threadIdx.x >> 5);
  if (wid >= (M >> 4) * ntiles) return;
  const int mBase = (wid / ntiles) * 16;
  const int nBase = (wid % ntiles) * 16;

  const _Float16* Arow = A + (size_t)(mBase + l15) * lda + hi * 8;
  const float*    Wrow = W + (size_t)(nBase + l15) * ldw + hi * 8;
  const int Kmain = K & ~31;

  v8f acc = {};
  for (int k0 = 0; k0 < Kmain; k0 += 32) {
    if ((k0 & 255) == 0)
      __builtin_prefetch(Wrow + k0 + 1024, 0, 1);   // global_prefetch_b8
    const v8h alo = *(const v8h*)(Arow + k0);
    const v8h ahi = *(const v8h*)(Arow + k0 + 16);
    v8h blo, bhi;
    #pragma unroll
    for (int j = 0; j < 4; ++j) {
      const float2 w0 = *(const float2*)(Wrow + k0 + 2 * j);
      const float2 w1 = *(const float2*)(Wrow + k0 + 16 + 2 * j);
      blo[2*j]   = (_Float16)w0.x; blo[2*j+1] = (_Float16)w0.y;
      bhi[2*j]   = (_Float16)w1.x; bhi[2*j+1] = (_Float16)w1.y;
    }
    acc = __builtin_amdgcn_wmma_f32_16x16x32_f16(
        false, frag16(alo, ahi), false, frag16(blo, bhi),
        (short)0, acc, false, false);
  }
  if (Kmain < K) {                     // single guarded tail slice
    v16h a, bm;
    #pragma unroll
    for (int i = 0; i < 16; ++i) {
      const int k = Kmain + ((i < 8) ? i : i + 8);
      _Float16 av = (_Float16)0.f, bv = (_Float16)0.f;
      if (k + hi * 8 < K) { av = Arow[k]; bv = (_Float16)Wrow[k]; }
      a[i] = av; bm[i] = bv;
    }
    acc = __builtin_amdgcn_wmma_f32_16x16x32_f16(false, a, false, bm,
                                                 (short)0, acc, false, false);
  }

  #pragma unroll
  for (int j = 0; j < 8; ++j) {
    const int m = mBase + j + (hi ? 8 : 0);
    const int n = nBase + l15;
    float v = acc[j] + bias[n];
    if (doLeaky) v = leakyf(v);
    outH[(size_t)m * N + n] = (_Float16)v;
  }
}

// Tiny head GEMM: out[b,o] = act[b,:Kd].w[o,:Kd] + bias[o], o in [0,4).
__global__ void head4_kernel(const _Float16* __restrict__ act,
                             const float* __restrict__ w,
                             const float* __restrict__ bias,
                             float* __restrict__ out, int Kd) {
  const int tid = blockIdx.x * blockDim.x + threadIdx.x;
  if (tid >= NB * 4) return;
  const int b = tid >> 2, o = tid & 3;
  const _Float16* a = act + (size_t)b * Kd;
  const float*    wr = w + (size_t)o * Kd;
  float acc = 0.f;
  #pragma unroll 8
  for (int k = 0; k < Kd; ++k) acc += (float)a[k] * wr[k];
  out[b * 4 + o] = acc + bias[o];
}

// Normalize quats, quat_dist vs targets, mean over batch -> d_out[0..1].
__global__ void loss_kernel(const float* __restrict__ lo,
                            const float* __restrict__ ro,
                            const float* __restrict__ lt,
                            const float* __restrict__ rt,
                            float* __restrict__ out) {
  __shared__ float sl[NB], sr[NB];
  const int b = threadIdx.x;
  float d2[2];
  const float* qs[2] = { lo + b * 4, ro + b * 4 };
  const float* ts[2] = { lt + b * 4, rt + b * 4 };
  #pragma unroll
  for (int s = 0; s < 2; ++s) {
    const float* q4 = qs[s]; const float* t4 = ts[s];
    float n = sqrtf(q4[0]*q4[0] + q4[1]*q4[1] + q4[2]*q4[2] + q4[3]*q4[3]);
    n = fmaxf(n, 1e-12f);
    float qw = q4[0]/n, qx = q4[1]/n, qy = q4[2]/n, qz = q4[3]/n;
    float rw = t4[0], rx = -t4[1], ry = -t4[2], rz = -t4[3];
    float tw = qw*rw - qx*rx - qy*ry - qz*rz;
    float tx = qw*rx + qx*rw + qy*rz - qz*ry;
    float ty = qw*ry - qx*rz + qy*rw + qz*rx;
    float tz = qw*rz + qx*ry - qy*rx + qz*rw;
    d2[s] = 2.f * atan2f(sqrtf(tx*tx + ty*ty + tz*tz + 1e-12f), fabsf(tw));
  }
  sl[b] = d2[0]; sr[b] = d2[1];
  __syncthreads();
  for (int s = NB / 2; s > 0; s >>= 1) {
    if (b < s) { sl[b] += sl[b + s]; sr[b] += sr[b + s]; }
    __syncthreads();
  }
  if (b == 0) { out[0] = sl[0] / (float)NB; out[1] = sr[0] / (float)NB; }
}

extern "C" void kernel_launch(void* const* d_in, const int* in_sizes, int n_in,
                              void* d_out, int out_size, void* d_ws, size_t ws_size,
                              hipStream_t stream) {
  const float* left  = (const float*)d_in[0];
  const float* right = (const float*)d_in[1];
  const float* lt    = (const float*)d_in[2];
  const float* rt    = (const float*)d_in[3];
  const float* wc[5] = { (const float*)d_in[4], (const float*)d_in[6],
                         (const float*)d_in[8], (const float*)d_in[10],
                         (const float*)d_in[12] };
  const float* bc[5] = { (const float*)d_in[5], (const float*)d_in[7],
                         (const float*)d_in[9], (const float*)d_in[11],
                         (const float*)d_in[13] };
  const float* wfc1 = (const float*)d_in[14]; const float* bfc1 = (const float*)d_in[15];
  const float* wl1  = (const float*)d_in[16]; const float* bl1  = (const float*)d_in[17];
  const float* wr1  = (const float*)d_in[18]; const float* br1  = (const float*)d_in[19];
  const float* wl2  = (const float*)d_in[20]; const float* bl2  = (const float*)d_in[21];
  const float* wr2  = (const float*)d_in[22]; const float* br2  = (const float*)d_in[23];
  float* out = (float*)d_out;

  static const int OC[5]    = {128, 128, 96, 64, 32};
  static const int IC[5]    = { 81, 209, 337, 433, 497};
  static const int ICP[5]   = { 96, 224, 352, 448, 512};   // == CHP - inOff
  static const int INOFF[5] = {448, 320, 192,  96,  32};
  static const int OUTOFF[5]= {320, 192,  96,  32,   0};

  char* ws = (char*)d_ws;
  size_t off = 0;
  _Float16* xN = (_Float16*)(ws + off); off += (size_t)NB * NSTRIDE * 2;   // 50.1 MB
  _Float16* xC = (_Float16*)(ws + off); off += (size_t)NB * KFCP * 2;      // 32.2 MB
  _Float16* wpack[5];
  for (int i = 0; i < 5; ++i) { wpack[i] = (_Float16*)(ws + off); off += (size_t)9 * OC[i] * ICP[i] * 2; }
  off = (off + 255) & ~(size_t)255;
  _Float16* fc1H = (_Float16*)(ws + off); off += (size_t)NB * 512 * 2;
  _Float16* loH  = (_Float16*)(ws + off); off += (size_t)NB * 256 * 2;
  _Float16* roH  = (_Float16*)(ws + off); off += (size_t)NB * 256 * 2;
  float* lo2 = (float*)(ws + off); off += (size_t)NB * 4 * 4;
  float* ro2 = (float*)(ws + off);

  // 0) zero NHWC (halo + pad channels), repack weights, zero NCHW tail pad
  {
    const size_t total16 = (size_t)NB * NSTRIDE / 8;
    zeroN_kernel<<<(int)((total16 + 255) / 256), 256, 0, stream>>>(xN);
  }
  for (int i = 0; i < 5; ++i) {
    const int total = 9 * OC[i] * ICP[i];
    pack_w_kernel<<<(total + 255) / 256, 256, 0, stream>>>(wc[i], wpack[i], OC[i], IC[i], ICP[i]);
  }
  padC_kernel<<<(NB * (KFCP - KFC) + 255) / 256, 256, 0, stream>>>(xC);

  // 1) correlation volume -> both layouts
  corr81_kernel<<<NB * 81, 256, 0, stream>>>(left, right, xN, xC);

  // 2) dense-net conv stack (double-buffered async weight staging + WMMA)
  for (int i = 0; i < 5; ++i) {
    const int blocks = NB * (OC[i] >> 4) * 2;
    conv3_wmma_kernel<<<blocks, 256, 0, stream>>>(
        xN, xC, wpack[i], bc[i], OC[i], INOFF[i], ICP[i], OUTOFF[i]);
  }

  // 3) fc1: (128 x 125902) @ (125902 x 512), leaky; xC is the flattened A.
  fc_wmma_kernel<<<32, 256, 0, stream>>>(xC, KFCP, wfc1, KFC, bfc1, fc1H, NB, 512, KFC, 1);
  // 4) l1 / r1: (128 x 512) @ (512 x 256), leaky
  fc_wmma_kernel<<<16, 256, 0, stream>>>(fc1H, 512, wl1, 512, bl1, loH, NB, 256, 512, 1);
  fc_wmma_kernel<<<16, 256, 0, stream>>>(fc1H, 512, wr1, 512, br1, roH, NB, 256, 512, 1);

  // 5) l2 / r2 heads (N=4, VALU)
  head4_kernel<<<2, 256, 0, stream>>>(loH, wl2, bl2, lo2, 256);
  head4_kernel<<<2, 256, 0, stream>>>(roH, wr2, br2, ro2, 256);

  // 6) quaternion geodesic loss, batch mean
  loss_kernel<<<1, NB, 0, stream>>>(lo2, ro2, lt, rt, out);
}
